// ThermoMHA_10892037063074
// MI455X (gfx1250) — matrix operand order repaired and checked by reading, not compiled, not run on hardware
//
#include <hip/hip_runtime.h>
#include <hip/hip_bf16.h>
#include <stddef.h>

// ---------------------------------------------------------------------------
// ThermoMHA on MI455X (gfx1250): bf16 WMMA, fp32 accum/softmax.
// B=2, L=2048, D=1024, H=16, Dh=64.  Outputs: out [B,L,D] ++ w [B,H,L,L].
// ---------------------------------------------------------------------------

typedef __bf16 bf16_t;
typedef __attribute__((ext_vector_type(16))) __bf16 v16bf;
typedef __attribute__((ext_vector_type(8)))  __bf16 v8bf;
typedef __attribute__((ext_vector_type(8)))  float  v8f;
typedef __attribute__((ext_vector_type(4)))  int    v4i;

#define Bsz   2
#define Lsz   2048
#define Dsz   1024
#define Hn    16
#define Dh    64
#define NEGV  (-1000000000.0f)
#define LDP   2052   // padded LDS row stride (floats): 16 rows hit distinct banks

// gfx1250 async LDS->global store path (drains w rows via ASYNCcnt engine,
// overlapping the P@V WMMAs). Signature probed via compile diagnostics:
//   (AS1 v4i* global_dst, AS3 v4i* lds_src, imm i32 offset, imm i32 cpol)
#if defined(__HIP_DEVICE_COMPILE__) && \
    __has_builtin(__builtin_amdgcn_global_store_async_from_lds_b128)
#define USE_ASYNC_W 1
typedef __attribute__((address_space(1))) v4i* as1_v4i;
typedef __attribute__((address_space(3))) v4i* as3_v4i;
#else
#define USE_ASYNC_W 0
#endif

// ---- WMMA wrapper: D = A(16x32 bf16) * B(32x16 bf16) + C(16x16 f32) -------
__device__ __forceinline__ v8f wmma_bf16(v16bf a, v16bf b, v8f c) {
  return __builtin_amdgcn_wmma_f32_16x16x32_bf16(
      /*neg_a=*/false, a, /*neg_b=*/false, b,
      /*c_mod=*/(short)0, c, /*reuse_a=*/false, /*reuse_b=*/false);
}

// ---- A fragment: rows row0..row0+15 of row-major [.,ld] bf16, K = k0..k0+31.
// Contiguous K order (matches B): lane<16 -> row=lane, K {k0..k0+15};
// lane>=16 -> row=lane-16, K {k0+16..k0+31}. One 32B load.
__device__ __forceinline__ v16bf load_a_frag(const bf16_t* base, int ld,
                                             int row0, int k0, int lane) {
  int row = row0 + (lane & 15);
  int kb  = k0 + ((lane >> 4) << 4);
  return *(const v16bf*)(base + (size_t)row * ld + kb);
}

// ---- B fragment: B is KxN; we store B^T row-major so column n of B is a
// contiguous row of B^T.  Contiguous K order (matches A): lane<16 -> col=lane,
// K {k0..k0+15}; lane>=16 -> col=lane-16, K {k0+16..k0+31}. One 32B load.
__device__ __forceinline__ v16bf load_b_frag(const bf16_t* baseT, int ld,
                                             int col0, int k0, int lane) {
  int col = col0 + (lane & 15);
  int kb  = k0 + ((lane >> 4) << 4);
  return *(const v16bf*)(baseT + (size_t)col * ld + kb);
}

// ===========================================================================
// Kernel 1: casts + transposes.
//   x[BL,D] f32 -> x_bf;  Wqkv[D,3D] -> WqkvT[3D,D] bf16;  Wo[D,D] -> WoT bf16
// ===========================================================================
__global__ void prep_kernel(const float* __restrict__ x,
                            const float* __restrict__ Wqkv,
                            const float* __restrict__ Wo,
                            bf16_t* __restrict__ x_bf,
                            bf16_t* __restrict__ WqkvT,
                            bf16_t* __restrict__ WoT) {
  const int NX = Bsz * Lsz * Dsz;        // 4194304
  const int NW1 = 3 * Dsz * Dsz;         // 3145728
  const int NW2 = Dsz * Dsz;             // 1048576
  int stride = gridDim.x * blockDim.x;
  for (int i = blockIdx.x * blockDim.x + threadIdx.x; i < NX + NW1 + NW2;
       i += stride) {
    if (i < NX) {
      x_bf[i] = (bf16_t)x[i];
    } else if (i < NX + NW1) {
      int j = i - NX;
      int n = j >> 10, k = j & 1023;              // WqkvT[n][k] = Wqkv[k][n]
      WqkvT[j] = (bf16_t)Wqkv[(size_t)k * (3 * Dsz) + n];
    } else {
      int j = i - NX - NW1;
      int n = j >> 10, k = j & 1023;              // WoT[n][k] = Wo[k][n]
      WoT[j] = (bf16_t)Wo[(size_t)k * Dsz + n];
    }
  }
}

// ===========================================================================
// Kernel 2: qkv = x_bf @ Wqkv  (M=4096, N=3072, K=1024), wave per 64x64 tile.
// Scatter into Qb[b,h,l,dh], Kb[b,h,l,dh], VT[b,h,dh,l] (bf16).
// ===========================================================================
__global__ __launch_bounds__(128) void qkv_gemm_kernel(
    const bf16_t* __restrict__ A, const bf16_t* __restrict__ BT,
    bf16_t* __restrict__ Qb, bf16_t* __restrict__ Kb, bf16_t* __restrict__ VT) {
  int lane = threadIdx.x & 31;
  int wave = threadIdx.x >> 5;
  int tile = blockIdx.x * 4 + wave;      // 3072 tiles = 64 Mtiles x 48 Ntiles
  int m0 = (tile / 48) * 64;
  int n0 = (tile % 48) * 64;

  v8f acc[4][4];
#pragma unroll
  for (int mi = 0; mi < 4; ++mi)
#pragma unroll
    for (int ni = 0; ni < 4; ++ni) acc[mi][ni] = (v8f)0.0f;

  for (int k0 = 0; k0 < Dsz; k0 += 32) {
    // speculative prefetch of the next A k-slab (translation faults dropped)
    __builtin_prefetch(A + (size_t)(m0 + (lane & 15)) * Dsz + k0 + 32, 0, 1);
    v16bf af[4], bfr[4];
#pragma unroll
    for (int mi = 0; mi < 4; ++mi)
      af[mi] = load_a_frag(A, Dsz, m0 + 16 * mi, k0, lane);
#pragma unroll
    for (int ni = 0; ni < 4; ++ni)
      bfr[ni] = load_b_frag(BT, Dsz, n0 + 16 * ni, k0, lane);
#pragma unroll
    for (int mi = 0; mi < 4; ++mi)
#pragma unroll
      for (int ni = 0; ni < 4; ++ni)
        acc[mi][ni] = wmma_bf16(af[mi], bfr[ni], acc[mi][ni]);
  }

  // scatter C (f32) -> bf16 head-major buffers
#pragma unroll
  for (int mi = 0; mi < 4; ++mi) {
#pragma unroll
    for (int ni = 0; ni < 4; ++ni) {
#pragma unroll
      for (int i = 0; i < 8; ++i) {
        int m = m0 + 16 * mi + i + ((lane >> 4) << 3);
        int n = n0 + 16 * ni + (lane & 15);
        int b = m >> 11;            // /2048
        int lpos = m & 2047;
        int which = n >> 10;        // 0=q 1=k 2=v
        int hn = n & 1023;
        int h = hn >> 6, dh = hn & 63;
        int bh = b * Hn + h;
        bf16_t v = (bf16_t)acc[mi][ni][i];
        if (which == 0)
          Qb[((size_t)bh * Lsz + lpos) * Dh + dh] = v;
        else if (which == 1)
          Kb[((size_t)bh * Lsz + lpos) * Dh + dh] = v;
        else
          VT[((size_t)bh * Dh + dh) * Lsz + lpos] = v;   // V transposed
      }
    }
  }
}

// ===========================================================================
// Kernel 3: attention. One 128-thread WG per (b,h,qtile of 16 rows).
// LDS: P[16][LDP] f32 (128KB+pad).  S=QK^T/(8*tau) masked -> softmax ->
// write w once (async LDS->global when available) -> O = P @ V via WMMA.
// ===========================================================================
__global__ __launch_bounds__(128) void attn_kernel(
    const bf16_t* __restrict__ Qb, const bf16_t* __restrict__ Kb,
    const bf16_t* __restrict__ VT, const float* __restrict__ tau,
    float* __restrict__ w_out, bf16_t* __restrict__ Ob) {
  extern __shared__ float P[];   // [16][LDP]
  int lane = threadIdx.x & 31;
  int wave = threadIdx.x >> 5;
  int qt = blockIdx.x & 127;           // L/16
  int h  = (blockIdx.x >> 7) & 15;
  int b  = blockIdx.x >> 11;
  int bh = b * Hn + h;
  const bf16_t* Q = Qb + (size_t)bh * Lsz * Dh;
  const bf16_t* K = Kb + (size_t)bh * Lsz * Dh;
  const bf16_t* V = VT + (size_t)bh * Dh * Lsz;
  int l0 = qt * 16;
  int limit = l0 + 16;                 // cols [0,limit) are live (causal)
  float scale = 1.0f / (8.0f * tau[bh]);   // 1/(sqrt(64)*tau)

  // ---- pass 1: score tiles (waves split key-tiles round-robin) ----
  v16bf qa0 = load_a_frag(Q, Dh, l0, 0, lane);
  v16bf qa1 = load_a_frag(Q, Dh, l0, 32, lane);
  for (int kt = wave; kt <= qt; kt += 4) {
    v16bf kb0 = load_b_frag(K, Dh, kt * 16, 0, lane);   // B = K^T
    v16bf kb1 = load_b_frag(K, Dh, kt * 16, 32, lane);
    v8f s = (v8f)0.0f;
    s = wmma_bf16(qa0, kb0, s);
    s = wmma_bf16(qa1, kb1, s);
    int col = kt * 16 + (lane & 15);
#pragma unroll
    for (int i = 0; i < 8; ++i) {
      int row = i + ((lane >> 4) << 3);
      float v = s[i] * scale;
      if (col > l0 + row) v = NEGV;                     // causal mask
      P[row * LDP + col] = v;
    }
  }
  __syncthreads();

  // ---- pass 2: fp32 softmax per row (wave owns 4 rows); emit w row ----
  for (int r = wave * 4; r < wave * 4 + 4; ++r) {
    float* prow = P + r * LDP;
    float m = -3.0e38f;
    for (int c = lane; c < limit; c += 32) m = fmaxf(m, prow[c]);
#pragma unroll
    for (int d = 16; d > 0; d >>= 1) m = fmaxf(m, __shfl_xor(m, d, 32));
    float ssum = 0.0f;
    for (int c = lane; c < limit; c += 32) {
      float e = __expf(prow[c] - m);
      prow[c] = e;
      ssum += e;
    }
#pragma unroll
    for (int d = 16; d > 0; d >>= 1) ssum += __shfl_xor(ssum, d, 32);
    float inv = 1.0f / ssum;
    // normalize in LDS (pass 3 consumes P; w row is a copy of it)
    for (int c = lane; c < limit; c += 32) prow[c] *= inv;

    float* wrow = w_out + ((size_t)bh * Lsz + (l0 + r)) * Lsz;
#if USE_ASYNC_W
    // order this wave's LDS normalize stores before the async engine reads
    asm volatile("s_wait_dscnt 0x0" ::: "memory");
    // 16B chunks; limit is a multiple of 16 floats, bases are 16B aligned
    for (int c = lane * 4; c < limit; c += 128)
      __builtin_amdgcn_global_store_async_from_lds_b128(
          (as1_v4i)(wrow + c), (as3_v4i)(prow + c), 0, 0);
#else
    for (int c = lane; c < limit; c += 32)
      __builtin_nontemporal_store(prow[c], wrow + c);
#endif
    // zeros above the causal limit (streaming, never re-read -> NT)
    for (int c = limit + lane; c < Lsz; c += 32)
      __builtin_nontemporal_store(0.0f, wrow + c);
  }
  __syncthreads();

  // ---- pass 3: O(16 x 64) = P(16 x limit) @ V(limit x 64); wave -> 16 dh ----
  v8f o = (v8f)0.0f;
  int msteps = (limit + 31) >> 5;
  for (int ms = 0; ms < msteps; ++ms) {
    int k0 = ms * 32;
    int arow = lane & 15;
    int kb = k0 + ((lane >> 4) << 4);
    v16bf af;
#pragma unroll
    for (int i = 0; i < 16; ++i) {
      int ka = kb + i;
      af[i] = (ka < limit) ? (bf16_t)P[arow * LDP + ka] : (bf16_t)0.0f;
    }
    v16bf bfr = load_b_frag(V, Lsz, wave * 16, k0, lane);  // V^T rows = V cols
    o = wmma_bf16(af, bfr, o);
  }
#pragma unroll
  for (int i = 0; i < 8; ++i) {
    int row = i + ((lane >> 4) << 3);
    Ob[((size_t)(b * Lsz + l0 + row)) * Dsz + h * Dh + wave * 16 + (lane & 15)] =
        (bf16_t)o[i];
  }
  // (async w stores drain before S_ENDPGM's implicit wait-idle)
}

// ===========================================================================
// Kernel 4: out = Ob @ Wo  (M=4096, N=1024, K=1024) -> f32 d_out
// ===========================================================================
__global__ __launch_bounds__(128) void out_gemm_kernel(
    const bf16_t* __restrict__ A, const bf16_t* __restrict__ BT,
    float* __restrict__ out) {
  int lane = threadIdx.x & 31;
  int wave = threadIdx.x >> 5;
  int tile = blockIdx.x * 4 + wave;    // 1024 tiles = 64 Mtiles x 16 Ntiles
  int m0 = (tile >> 4) * 64;
  int n0 = (tile & 15) * 64;

  v8f acc[4][4];
#pragma unroll
  for (int mi = 0; mi < 4; ++mi)
#pragma unroll
    for (int ni = 0; ni < 4; ++ni) acc[mi][ni] = (v8f)0.0f;

  for (int k0 = 0; k0 < Dsz; k0 += 32) {
    __builtin_prefetch(A + (size_t)(m0 + (lane & 15)) * Dsz + k0 + 32, 0, 1);
    v16bf af[4], bfr[4];
#pragma unroll
    for (int mi = 0; mi < 4; ++mi)
      af[mi] = load_a_frag(A, Dsz, m0 + 16 * mi, k0, lane);
#pragma unroll
    for (int ni = 0; ni < 4; ++ni)
      bfr[ni] = load_b_frag(BT, Dsz, n0 + 16 * ni, k0, lane);
#pragma unroll
    for (int mi = 0; mi < 4; ++mi)
#pragma unroll
      for (int ni = 0; ni < 4; ++ni)
        acc[mi][ni] = wmma_bf16(af[mi], bfr[ni], acc[mi][ni]);
  }

#pragma unroll
  for (int mi = 0; mi < 4; ++mi)
#pragma unroll
    for (int ni = 0; ni < 4; ++ni)
#pragma unroll
      for (int i = 0; i < 8; ++i) {
        int m = m0 + 16 * mi + i + ((lane >> 4) << 3);
        int n = n0 + 16 * ni + (lane & 15);
        out[(size_t)m * Dsz + n] = acc[mi][ni][i];
      }
}

// ===========================================================================
extern "C" void kernel_launch(void* const* d_in, const int* in_sizes, int n_in,
                              void* d_out, int out_size, void* d_ws,
                              size_t ws_size, hipStream_t stream) {
  (void)in_sizes; (void)n_in; (void)out_size; (void)ws_size;
  const float* x    = (const float*)d_in[0];
  const float* tau  = (const float*)d_in[1];
  const float* Wqkv = (const float*)d_in[2];
  const float* Wo   = (const float*)d_in[3];

  float* out = (float*)d_out;                       // [B,L,D] f32
  float* w   = (float*)d_out + (size_t)Bsz * Lsz * Dsz;   // [B,H,L,L] f32

  char* ws = (char*)d_ws;
  bf16_t* x_bf  = (bf16_t*)ws;                 ws += (size_t)Bsz*Lsz*Dsz*2;
  bf16_t* WqkvT = (bf16_t*)ws;                 ws += (size_t)3*Dsz*Dsz*2;
  bf16_t* WoT   = (bf16_t*)ws;                 ws += (size_t)Dsz*Dsz*2;
  bf16_t* Qb    = (bf16_t*)ws;                 ws += (size_t)Bsz*Lsz*Dsz*2;
  bf16_t* Kb    = (bf16_t*)ws;                 ws += (size_t)Bsz*Lsz*Dsz*2;
  bf16_t* VT    = (bf16_t*)ws;                 ws += (size_t)Bsz*Lsz*Dsz*2;
  bf16_t* Ob    = (bf16_t*)ws;

  // 1) casts/transposes
  prep_kernel<<<8192, 256, 0, stream>>>(x, Wqkv, Wo, x_bf, WqkvT, WoT);

  // 2) QKV projection: 3072 wave-tiles / 4 waves per block
  qkv_gemm_kernel<<<768, 128, 0, stream>>>(x_bf, WqkvT, Qb, Kb, VT);

  // 3) attention: B*H*(L/16) = 4096 workgroups, 128KB dynamic LDS each
  const int attn_lds = 16 * LDP * sizeof(float);    // 131328 B (<320KB/WGP)
  (void)hipFuncSetAttribute((const void*)attn_kernel,
                            hipFuncAttributeMaxDynamicSharedMemorySize,
                            attn_lds);
  attn_kernel<<<Bsz * Hn * (Lsz / 16), 128, attn_lds, stream>>>(
      Qb, Kb, VT, tau, w, Ob);

  // 4) output projection: 1024 wave-tiles / 4 waves per block
  out_gemm_kernel<<<256, 128, 0, stream>>>(Ob, WoT, out);
}
